// FDModel_9079560863859
// MI455X (gfx1250) — matrix operand order repaired and checked by reading, not compiled, not run on hardware
//
#include <hip/hip_runtime.h>
#include <hip/hip_bf16.h>

#define NROWS 6144
#define TOPK 5

typedef __attribute__((ext_vector_type(16))) __bf16 v16bf;
typedef __attribute__((ext_vector_type(8)))  __bf16 v8bf;
typedef __attribute__((ext_vector_type(8)))  float  v8f;
typedef __attribute__((ext_vector_type(2)))  float  v2f;

// Concatenate two 16-byte chunks into one 32-element-pair WMMA fragment.
__device__ __forceinline__ v16bf frag_cat(const __bf16* p0, const __bf16* p1) {
  v8bf a = *(const v8bf*)p0;
  v8bf b = *(const v8bf*)p1;
  v16bf r;
#pragma unroll
  for (int i = 0; i < 8; ++i) { r[i] = a[i]; r[i + 8] = b[i]; }
  return r;
}

__device__ __forceinline__ void topk_insert(float (&tv)[TOPK], int (&ti)[TOPK],
                                            float v, int idx) {
  if (v > tv[TOPK - 1]) {
    tv[TOPK - 1] = v; ti[TOPK - 1] = idx;
#pragma unroll
    for (int j = TOPK - 1; j > 0; --j) {
      if (tv[j] > tv[j - 1]) {
        float t = tv[j]; tv[j] = tv[j - 1]; tv[j - 1] = t;
        int u = ti[j]; ti[j] = ti[j - 1]; ti[j - 1] = u;
      }
    }
  }
}

// Fast sigmoid using v_rcp_f32 (1-ulp; reference tolerance is far looser).
__device__ __forceinline__ float fast_sigmoid(float z) {
  return __builtin_amdgcn_rcpf(1.f + __expf(-z));
}

// ---------------------------------------------------------------------------
// Kernel 1: row-normalize x and write bf16 hi/lo split (bf16x3 GEMM inputs).
// ---------------------------------------------------------------------------
template <int D>
__global__ __launch_bounds__(256) void normalize_split(
    const float* __restrict__ x, __bf16* __restrict__ xh, __bf16* __restrict__ xl) {
  const int row = blockIdx.x;
  const int tid = threadIdx.x;
  __shared__ float red[256];
  const float* xr = x + (size_t)row * D;
  float s = 0.f;
  for (int d = tid; d < D; d += 256) { float v = xr[d]; s = fmaf(v, v, s); }
  red[tid] = s;
  __syncthreads();
#pragma unroll
  for (int off = 128; off > 0; off >>= 1) {
    if (tid < off) red[tid] += red[tid + off];
    __syncthreads();
  }
  const float inv = 1.f / fmaxf(sqrtf(red[0]), 1e-12f);
  __bf16* hr = xh + (size_t)row * D;
  __bf16* lr = xl + (size_t)row * D;
  for (int d = tid; d < D; d += 256) {
    float q = xr[d] * inv;
    __bf16 h = (__bf16)q;
    hr[d] = h;
    lr[d] = (__bf16)(q - (float)h);
  }
}

// ---------------------------------------------------------------------------
// Kernel 2: fused Gram (bf16x3 WMMA) + masked streaming top-5 + imputation.
// Block = 8 wave32s, owns a 16-row A tile in LDS; waves sweep column tiles.
// ---------------------------------------------------------------------------
template <int D>
__global__ __launch_bounds__(256) void gram_impute(
    const float* __restrict__ xin,
    const __bf16* __restrict__ xqh,
    const __bf16* __restrict__ xql,
    const int* __restrict__ maskv,   // pre-offset by view; stride 3
    float* __restrict__ out) {
  __shared__ __align__(16) __bf16 sAh[16 * D];
  __shared__ __align__(16) __bf16 sAl[16 * D];
  __shared__ float sScore[8][16][17];
  __shared__ float sPV[16][16][TOPK];
  __shared__ int   sPI[16][16][TOPK];
  __shared__ float sW[16][TOPK];
  __shared__ int   sI[16][TOPK];

  const int tid  = threadIdx.x;
  const int wave = tid >> 5;
  const int lane = tid & 31;
  const int half = lane >> 4;
  const int nn   = lane & 15;
  const int rowbase = blockIdx.x * 16;

  // Cooperative load of the 16xD hi/lo A tile into LDS.
  {
    const unsigned int* gh = (const unsigned int*)(xqh + (size_t)rowbase * D);
    const unsigned int* gl = (const unsigned int*)(xql + (size_t)rowbase * D);
    unsigned int* lh = (unsigned int*)sAh;
    unsigned int* ll = (unsigned int*)sAl;
    for (int i = tid; i < 8 * D; i += 256) { lh[i] = gh[i]; ll[i] = gl[i]; }
  }
  __syncthreads();

  float tv[TOPK]; int ti[TOPK];
#pragma unroll
  for (int k = 0; k < TOPK; ++k) { tv[k] = -__builtin_inff(); ti[k] = 0; }

  // A fragment sources (ISA 16-bit A layout: K chunks at 8*half and 16+8*half).
  const __bf16* ahh = sAh + (size_t)nn * D + 8 * half;
  const __bf16* all = sAl + (size_t)nn * D + 8 * half;

  for (int ct = wave; ct < NROWS / 16; ct += 8) {
    const int col = ct * 16 + nn;
    // B fragment source (same K order as A: chunks at 8*half and 16+8*half).
    const __bf16* bh = xqh + (size_t)col * D + 8 * half;
    const __bf16* bl = xql + (size_t)col * D + 8 * half;
    v8f acc = {};
#pragma unroll 4
    for (int kk = 0; kk < D; kk += 32) {
      v16bf Ah = frag_cat(ahh + kk, ahh + kk + 16);
      v16bf Al = frag_cat(all + kk, all + kk + 16);
      v16bf Bh = frag_cat(bh + kk, bh + kk + 16);
      v16bf Bl = frag_cat(bl + kk, bl + kk + 16);
      acc = __builtin_amdgcn_wmma_f32_16x16x32_bf16(false, Ah, false, Bh, (short)0, acc, false, false);
      acc = __builtin_amdgcn_wmma_f32_16x16x32_bf16(false, Ah, false, Bl, (short)0, acc, false, false);
      acc = __builtin_amdgcn_wmma_f32_16x16x32_bf16(false, Al, false, Bh, (short)0, acc, false, false);
    }
    // Gate masked-out columns to -inf (exp is monotone: top-k on raw cosines).
    const float gate = (maskv[(size_t)col * 3] != 0) ? 0.f : -__builtin_inff();
#pragma unroll
    for (int v = 0; v < 8; ++v)
      sScore[wave][8 * half + v][nn] = acc[v] + gate;   // C/D row = v + 8*half
    __builtin_amdgcn_wave_barrier();                    // same-wave LDS is in-order
#pragma unroll
    for (int c = 0; c < 8; ++c) {
      float sv = sScore[wave][nn][8 * half + c];
      topk_insert(tv, ti, sv, ct * 16 + 8 * half + c);
    }
    __builtin_amdgcn_wave_barrier();
  }

  __syncthreads();
  {
    const int list = wave * 2 + half;   // 16 partial lists per row
#pragma unroll
    for (int k = 0; k < TOPK; ++k) { sPV[nn][list][k] = tv[k]; sPI[nn][list][k] = ti[k]; }
  }
  __syncthreads();

  if (tid < 16) {
    float fv[TOPK]; int fi[TOPK];
#pragma unroll
    for (int k = 0; k < TOPK; ++k) { fv[k] = -__builtin_inff(); fi[k] = 0; }
    for (int l = 0; l < 16; ++l)
#pragma unroll
      for (int k = 0; k < TOPK; ++k)
        topk_insert(fv, fi, sPV[tid][l][k], sPI[tid][l][k]);
    float w[TOPK]; float ssum = 0.f;
#pragma unroll
    for (int k = 0; k < TOPK; ++k) { w[k] = __expf(fv[k] * 5.0f); ssum += w[k]; }  // exp(s/0.2)
    const float rdn = (ssum == 0.f) ? 1.f : __builtin_amdgcn_rcpf(ssum);
#pragma unroll
    for (int k = 0; k < TOPK; ++k) { sW[tid][k] = w[k] * rdn; sI[tid][k] = fi[k]; }
  }
  __syncthreads();

  // Epilogue: weighted gather for missing rows, straight copy otherwise.
  for (int r = 0; r < 16; ++r) {
    const int grow = rowbase + r;
    float* orow = out + (size_t)grow * D;
    const float* xrow = xin + (size_t)grow * D;
    if (maskv[(size_t)grow * 3] != 0) {
      for (int d = tid; d < D; d += 256) orow[d] = xrow[d];
    } else {
      const float w0 = sW[r][0], w1 = sW[r][1], w2 = sW[r][2], w3 = sW[r][3], w4 = sW[r][4];
      const float* p0 = xin + (size_t)sI[r][0] * D;
      const float* p1 = xin + (size_t)sI[r][1] * D;
      const float* p2 = xin + (size_t)sI[r][2] * D;
      const float* p3 = xin + (size_t)sI[r][3] * D;
      const float* p4 = xin + (size_t)sI[r][4] * D;
      for (int d = tid; d < D; d += 256) {
        float a = w0 * p0[d];
        a = fmaf(w1, p1[d], a);
        a = fmaf(w2, p2[d], a);
        a = fmaf(w3, p3[d], a);
        a = fmaf(w4, p4[d], a);
        orow[d] = a;
      }
    }
  }
}

// ---------------------------------------------------------------------------
// Kernel 3: y_n = sigmoid(y @ W_y^T + b), exact fp32 via V_WMMA_F32_16X16X4_F32.
// ---------------------------------------------------------------------------
#if defined(__has_builtin)
#if __has_builtin(__builtin_amdgcn_wmma_f32_16x16x4_f32)
#define HAVE_WMMA_F32X4 1
#endif
#endif

#ifdef HAVE_WMMA_F32X4
__global__ __launch_bounds__(256) void ymlp(
    const float* __restrict__ y, const float* __restrict__ W,
    const float* __restrict__ b, float* __restrict__ out) {
  const int tid  = threadIdx.x;
  const int wave = tid >> 5;
  const int lane = tid & 31;
  const int half = lane >> 4;
  const int nn   = lane & 15;
  const int tileId  = blockIdx.x * 8 + wave;
  const int rowTile = tileId >> 6;        // 1024/16 = 64 column tiles
  const int colTile = tileId & 63;
  // f32 A 16x4: M = lane&15, K = g + 2*half  -> contiguous 2-float load.
  const float* ya = y + (size_t)(rowTile * 16 + nn) * 256 + 2 * half;
  const float* wb = W + (size_t)(colTile * 16 + nn) * 256 + 2 * half;
  v8f acc = {};
#pragma unroll 8
  for (int kk = 0; kk < 256; kk += 4) {
    v2f a  = *(const v2f*)(ya + kk);
    v2f bb = *(const v2f*)(wb + kk);
    acc = __builtin_amdgcn_wmma_f32_16x16x4_f32(false, a, false, bb, (short)0, acc, false, false);
  }
  const float bias = b[colTile * 16 + nn];
#pragma unroll
  for (int v = 0; v < 8; ++v) {
    const int m = v + 8 * half;
    out[(size_t)(rowTile * 16 + m) * 1024 + colTile * 16 + nn] = fast_sigmoid(acc[v] + bias);
  }
}
#else
__global__ __launch_bounds__(256) void ymlp_fallback(
    const float* __restrict__ y, const float* __restrict__ W,
    const float* __restrict__ b, float* __restrict__ out) {
  const int idx = blockIdx.x * 256 + threadIdx.x;   // one output element each
  const int row = idx >> 10, col = idx & 1023;
  const float* yr = y + (size_t)row * 256;
  const float* wr = W + (size_t)col * 256;
  float s = b[col];
  for (int k = 0; k < 256; ++k) s = fmaf(yr[k], wr[k], s);
  out[(size_t)row * 1024 + col] = fast_sigmoid(s);
}
#endif

// ---------------------------------------------------------------------------
extern "C" void kernel_launch(void* const* d_in, const int* in_sizes, int n_in,
                              void* d_out, int out_size, void* d_ws, size_t ws_size,
                              hipStream_t stream) {
  (void)in_sizes; (void)n_in; (void)out_size; (void)ws_size;
  const float* x0  = (const float*)d_in[0];
  const float* x1  = (const float*)d_in[1];
  const float* x2  = (const float*)d_in[2];
  const float* y   = (const float*)d_in[3];
  const float* Wy  = (const float*)d_in[4];
  const float* by  = (const float*)d_in[5];
  const int*  mask = (const int*)d_in[6];
  float* out = (float*)d_out;

  const size_t n = NROWS;
  __bf16* h0 = (__bf16*)d_ws;          // bf16 hi/lo splits: 2 * n * 2304 elems = 56.6 MB
  __bf16* h1 = h0 + n * 512;
  __bf16* h2 = h1 + n * 768;
  __bf16* l0 = h2 + n * 1024;
  __bf16* l1 = l0 + n * 512;
  __bf16* l2 = l1 + n * 768;

  normalize_split<512 ><<<NROWS, 256, 0, stream>>>(x0, h0, l0);
  normalize_split<768 ><<<NROWS, 256, 0, stream>>>(x1, h1, l1);
  normalize_split<1024><<<NROWS, 256, 0, stream>>>(x2, h2, l2);

  gram_impute<512 ><<<NROWS / 16, 256, 0, stream>>>(x0, h0, l0, mask + 0, out);
  gram_impute<768 ><<<NROWS / 16, 256, 0, stream>>>(x1, h1, l1, mask + 1, out + n * 512);
  gram_impute<1024><<<NROWS / 16, 256, 0, stream>>>(x2, h2, l2, mask + 2, out + n * 1280);

#ifdef HAVE_WMMA_F32X4
  ymlp<<<(NROWS / 16) * 64 / 8, 256, 0, stream>>>(y, Wy, by, out + n * 2304);
#else
  ymlp_fallback<<<NROWS * 1024 / 256, 256, 0, stream>>>(y, Wy, by, out + n * 2304);
#endif
}
